// HiTransformer_4930622456126
// MI455X (gfx1250) — matrix operand order repaired and checked by reading, not compiled, not run on hardware
//
#include <hip/hip_runtime.h>
#include <math.h>

typedef __bf16 bf16;
typedef __attribute__((ext_vector_type(16))) __bf16 v16bf;
typedef __attribute__((ext_vector_type(8)))  float  v8f;

union FragU {
  uint4    q[2];
  unsigned u[8];
  v16bf    v;
};

static __device__ __forceinline__ unsigned pack2bf(float a, float b) {
  union { __bf16 h; unsigned short s; } ua, ub;
  ua.h = (__bf16)a; ub.h = (__bf16)b;
  return (unsigned)ua.s | ((unsigned)ub.s << 16);
}

#if defined(__has_builtin)
#if __has_builtin(__builtin_amdgcn_s_wait_asynccnt)
#define WAIT_ASYNC(n) __builtin_amdgcn_s_wait_asynccnt(n)
#endif
#endif
#ifndef WAIT_ASYNC
#define WAIT_ASYNC(n) asm volatile("s_wait_asynccnt %0" :: "n"(n) : "memory")
#endif

// Async global->LDS 16B copy (per-lane addresses). Tracked by ASYNCcnt.
static __device__ __forceinline__ void async_cp16(unsigned ldsOff, const void* gptr) {
  asm volatile("global_load_async_to_lds_b128 %0, %1, off" :: "v"(ldsOff), "v"(gptr) : "memory");
}
static __device__ __forceinline__ void async_cp16_off(unsigned ldsOff, const void* gptr, int imm) {
  if (imm == 16)
    asm volatile("global_load_async_to_lds_b128 %0, %1, off offset:16" :: "v"(ldsOff), "v"(gptr) : "memory");
  else if (imm == 32)
    asm volatile("global_load_async_to_lds_b128 %0, %1, off offset:32" :: "v"(ldsOff), "v"(gptr) : "memory");
  else
    asm volatile("global_load_async_to_lds_b128 %0, %1, off offset:48" :: "v"(ldsOff), "v"(gptr) : "memory");
}

// ---------------------------------------------------------------------------
// Tiled bf16 WMMA GEMM with double-buffered async-LDS staging.
//   C[M,N] = A[M,K](bf16) * B[K,N](fp32 -> bf16 at fragment build) + bias
// Block tile 128x128, K-step 32. 256 threads = 8 waves, each wave 2x4 tiles.
// Per K-step, per wave: 6 GLOBAL_LOAD_ASYNC_TO_LDS_B128 (2 for A, 4 for B).
// ---------------------------------------------------------------------------
template<bool RELU, bool OUTF, bool OUTB>
__global__ __launch_bounds__(256) void k_gemm(
    const bf16* __restrict__ A, const float* __restrict__ B,
    const float* __restrict__ bias, float* __restrict__ outF,
    bf16* __restrict__ outB, int M, int N, int K)
{
  __shared__ bf16  Ast[2][128][40];   // padded: conflict-free 128b frag reads
  __shared__ float Bst[2][32][132];   // raw fp32 weight tile, 16B-aligned rows

  const int tid  = threadIdx.x;
  const int lane = tid & 31;
  const int wave = tid >> 5;
  const int l16  = lane & 15;
  const int lh   = lane >> 4;
  const int bm   = blockIdx.y * 128;
  const int bn   = blockIdx.x * 128;
  const int wm   = (wave & 3) * 32;
  const int wn   = (wave >> 2) * 64;

  // async copy work assignment
  const int arow = tid >> 1;        // A: 128 rows, 2 threads/row
  const int acol = (tid & 1) * 2;   // 16B chunks {0,1} or {2,3}
  const int brow = tid >> 3;        // B: 32 rows, 8 threads/row
  const int bch  = (tid & 7) * 4;   // 4 x 16B chunks per thread

  const unsigned aLds0 = (unsigned)(size_t)&Ast[0][arow][0] + (unsigned)acol * 16u;
  const unsigned bLds0 = (unsigned)(size_t)&Bst[0][brow][0] + (unsigned)bch * 16u;
  const unsigned aBufSz = (unsigned)sizeof(Ast[0]);
  const unsigned bBufSz = (unsigned)sizeof(Bst[0]);

  const char* aG = (const char*)(A + (size_t)(bm + arow) * K) + acol * 16;
  const char* bG = (const char*)(B + (size_t)brow * N + bn) + bch * 16;
  const size_t aStep = 64;                  // 32 bf16 per K-step
  const size_t bStep = (size_t)32 * N * 4;  // 32 fp32 rows per K-step

  auto issue = [&](int t, int buf) {
    unsigned al = aLds0 + (unsigned)buf * aBufSz;
    const char* ag = aG + (size_t)t * aStep;
    async_cp16(al, ag);
    async_cp16_off(al, ag, 16);
    unsigned bl = bLds0 + (unsigned)buf * bBufSz;
    const char* bg = bG + (size_t)t * bStep;
    async_cp16(bl, bg);
    async_cp16_off(bl, bg, 16);
    async_cp16_off(bl, bg, 32);
    async_cp16_off(bl, bg, 48);
  };

  v8f acc[2][4];
#pragma unroll
  for (int i = 0; i < 2; ++i)
#pragma unroll
    for (int j = 0; j < 4; ++j)
      acc[i][j] = (v8f){0.f,0.f,0.f,0.f,0.f,0.f,0.f,0.f};

  const int nT = K >> 5;
  issue(0, 0);
  for (int t = 0; t < nT; ++t) {
    const int buf = t & 1;
    if (t + 1 < nT) {
      issue(t + 1, (t + 1) & 1);   // prefetch next tile into other buffer
      WAIT_ASYNC(6);               // in-order: tile t's 6 copies have landed
    } else {
      WAIT_ASYNC(0);
    }
    __syncthreads();

    FragU af[2];
#pragma unroll
    for (int mi = 0; mi < 2; ++mi) {
      const uint4* ap = reinterpret_cast<const uint4*>(&Ast[buf][wm + mi*16 + l16][0]);
      af[mi].q[0] = ap[2 * lh];       // k 0..7 / 16..23
      af[mi].q[1] = ap[2 * lh + 1];   // k 8..15 / 24..31
    }
    FragU bfr[4];
#pragma unroll
    for (int nj = 0; nj < 4; ++nj) {
      const int n = wn + nj*16 + l16;
#pragma unroll
      for (int vv = 0; vv < 8; ++vv) {
        const int p = lh * 8 + vv;                  // k-pair index
        float lo = Bst[buf][2*p][n];
        float hi = Bst[buf][2*p + 1][n];
        bfr[nj].u[vv] = pack2bf(lo, hi);
      }
    }
#pragma unroll
    for (int mi = 0; mi < 2; ++mi)
#pragma unroll
      for (int nj = 0; nj < 4; ++nj)
        acc[mi][nj] = __builtin_amdgcn_wmma_f32_16x16x32_bf16(
            false, af[mi].v, false, bfr[nj].v, (short)0, acc[mi][nj], false, false);
    __syncthreads();
  }

#pragma unroll
  for (int nj = 0; nj < 4; ++nj) {
    int gc = bn + wn + nj*16 + l16;
    float bv = bias ? bias[gc] : 0.f;
#pragma unroll
    for (int mi = 0; mi < 2; ++mi) {
#pragma unroll
      for (int r = 0; r < 8; ++r) {
        int gr = bm + wm + mi*16 + r + lh*8;
        float val = acc[mi][nj][r] + bv;
        if (RELU) val = fmaxf(val, 0.f);
        size_t idx = (size_t)gr * N + gc;
        if (OUTF) outF[idx] = val;
        if (OUTB) outB[idx] = (bf16)val;
      }
    }
  }
}

// ---------------------------------------------------------------------------
// Masked multi-head attention (flash-style, online softmax), one branch.
// Block = 128 threads (4 waves); wave = 16 query rows; grid = (L/64, NHEAD).
// ---------------------------------------------------------------------------
__global__ __launch_bounds__(128) void k_attn(
    const bf16* __restrict__ qkv, bf16* __restrict__ o,
    const int* __restrict__ utt, const int* __restrict__ spk,
    const int* __restrict__ winp, int branch)
{
  __shared__ bf16     Ks[64][72];      // k rows (padded stride)
  __shared__ unsigned Vt[64][33];      // [d][keypair] packed bf16
  __shared__ bf16     Ps[4][16][40];   // per-wave P staging (C->A relayout)

  const int tid   = threadIdx.x;
  const int lane  = tid & 31;
  const int wave  = tid >> 5;
  const int l16   = lane & 15;
  const int lh    = lane >> 4;
  const int head  = blockIdx.y;
  const int qbase = blockIdx.x * 64 + wave * 16;
  const int wrad  = *winp;
  const float scale = 0.125f;  // 1/sqrt(64)

  FragU qf0, qf1;
  {
    const uint4* qp = reinterpret_cast<const uint4*>(
        qkv + (size_t)(qbase + l16) * 3072 + head * 64);
    qf0.q[0] = qp[2 * lh];     qf0.q[1] = qp[2 * lh + 1];   // d 0..31
    qf1.q[0] = qp[4 + 2 * lh]; qf1.q[1] = qp[5 + 2 * lh];   // d 32..63
  }

  int spk_i[8], utt_i[8];
#pragma unroll
  for (int r = 0; r < 8; ++r) {
    int i = qbase + r + lh * 8;
    spk_i[r] = spk[i]; utt_i[r] = utt[i];
  }

  float mrow[8], lrow[8];
#pragma unroll
  for (int r = 0; r < 8; ++r) { mrow[r] = -1e30f; lrow[r] = 0.f; }
  v8f Oacc[4];
#pragma unroll
  for (int nj = 0; nj < 4; ++nj)
    Oacc[nj] = (v8f){0.f,0.f,0.f,0.f,0.f,0.f,0.f,0.f};

  for (int kc = 0; kc < 1024; kc += 64) {
    { // K tile: direct copy of k rows
      int row = tid >> 1, half = tid & 1;
      const uint4* s = reinterpret_cast<const uint4*>(
          qkv + (size_t)(kc + row) * 3072 + 1024 + head * 64) + half * 4;
      uint4* d = reinterpret_cast<uint4*>(&Ks[row][0]) + half * 4;
#pragma unroll
      for (int cc = 0; cc < 4; ++cc) d[cc] = s[cc];
    }
    { // V tile: transpose + pair-pack
      int p  = tid & 31;
      int d0 = (tid >> 5) * 16;
      const uint4* r0 = reinterpret_cast<const uint4*>(
          qkv + (size_t)(kc + 2*p) * 3072 + 2048 + head * 64 + d0);
      const uint4* r1 = reinterpret_cast<const uint4*>(
          qkv + (size_t)(kc + 2*p + 1) * 3072 + 2048 + head * 64 + d0);
      uint4 x0 = r0[0], x1 = r0[1], y0 = r1[0], y1 = r1[1];
      unsigned w0[8] = {x0.x,x0.y,x0.z,x0.w,x1.x,x1.y,x1.z,x1.w};
      unsigned w1[8] = {y0.x,y0.y,y0.z,y0.w,y1.x,y1.y,y1.z,y1.w};
#pragma unroll
      for (int jj = 0; jj < 8; ++jj) {
        Vt[d0 + 2*jj][p]     = (w0[jj] & 0xffffu) | (w1[jj] << 16);
        Vt[d0 + 2*jj + 1][p] = (w0[jj] >> 16) | (w1[jj] & 0xffff0000u);
      }
    }
    __syncthreads();

    for (int sub = 0; sub < 2; ++sub) {
      v8f S[2];
#pragma unroll
      for (int t2 = 0; t2 < 2; ++t2) {
        int keyl = sub * 32 + t2 * 16 + l16;
        const unsigned* kr = reinterpret_cast<const unsigned*>(&Ks[keyl][0]);
        FragU kf0, kf1;
#pragma unroll
        for (int vv = 0; vv < 8; ++vv) {
          kf0.u[vv] = kr[lh * 8 + vv];
          kf1.u[vv] = kr[16 + lh * 8 + vv];
        }
        v8f z = (v8f){0.f,0.f,0.f,0.f,0.f,0.f,0.f,0.f};
        S[t2] = __builtin_amdgcn_wmma_f32_16x16x32_bf16(false, qf0.v, false, kf0.v, (short)0, z, false, false);
        S[t2] = __builtin_amdgcn_wmma_f32_16x16x32_bf16(false, qf1.v, false, kf1.v, (short)0, S[t2], false, false);
      }
      float pv[2][8];
#pragma unroll
      for (int t2 = 0; t2 < 2; ++t2) {
        int j  = kc + sub * 32 + t2 * 16 + l16;
        int sj = spk[j], uj = utt[j];
#pragma unroll
        for (int r = 0; r < 8; ++r) {
          int i   = qbase + r + lh * 8;
          int dji = j - i;
          bool win   = (dji <= wrad) && (dji >= -wrad);
          bool valid = (uj > 0) && (utt_i[r] > 0);
          bool bas   = win && valid;
          bool same  = (sj == spk_i[r]);
          bool att;
          if (branch == 0)      att = bas;
          else if (branch == 1) att = bas && same;
          else                  att = bas && (!same || (i == j));
          pv[t2][r] = S[t2][r] * scale + (att ? 0.f : -1e9f);
        }
      }
#pragma unroll
      for (int r = 0; r < 8; ++r) {
        float mx = fmaxf(pv[0][r], pv[1][r]);
        mx = fmaxf(mx, __shfl_xor(mx, 1));
        mx = fmaxf(mx, __shfl_xor(mx, 2));
        mx = fmaxf(mx, __shfl_xor(mx, 4));
        mx = fmaxf(mx, __shfl_xor(mx, 8));
        float mn = fmaxf(mrow[r], mx);
        float c  = __expf(mrow[r] - mn);
        float p0 = __expf(pv[0][r] - mn);
        float p1 = __expf(pv[1][r] - mn);
        pv[0][r] = p0; pv[1][r] = p1;
        float rs = p0 + p1;
        rs += __shfl_xor(rs, 1);
        rs += __shfl_xor(rs, 2);
        rs += __shfl_xor(rs, 4);
        rs += __shfl_xor(rs, 8);
        lrow[r] = lrow[r] * c + rs;
        mrow[r] = mn;
#pragma unroll
        for (int nj = 0; nj < 4; ++nj) Oacc[nj][r] *= c;
      }
      // stage P (C layout) -> A-fragment layout via wave-private LDS
#pragma unroll
      for (int t2 = 0; t2 < 2; ++t2)
#pragma unroll
        for (int r = 0; r < 8; ++r)
          Ps[wave][r + lh * 8][t2 * 16 + l16] = (bf16)pv[t2][r];
      FragU pf;
      {
        const uint4* pp = reinterpret_cast<const uint4*>(&Ps[wave][l16][0]);
        pf.q[0] = pp[2 * lh];
        pf.q[1] = pp[2 * lh + 1];
      }
#pragma unroll
      for (int nj = 0; nj < 4; ++nj) {
        const unsigned* vp = &Vt[nj * 16 + l16][sub * 16 + lh * 8];
        FragU vf;
#pragma unroll
        for (int vv = 0; vv < 8; ++vv) vf.u[vv] = vp[vv];
        Oacc[nj] = __builtin_amdgcn_wmma_f32_16x16x32_bf16(
            false, pf.v, false, vf.v, (short)0, Oacc[nj], false, false);
      }
    }
    __syncthreads();
  }

#pragma unroll
  for (int r = 0; r < 8; ++r) {
    float inv = 1.f / lrow[r];
    int row = qbase + r + lh * 8;
#pragma unroll
    for (int nj = 0; nj < 4; ++nj)
      o[(size_t)row * 1024 + head * 64 + nj * 16 + l16] = (bf16)(Oacc[nj][r] * inv);
  }
}

// ---------------------------------------------------------------------------
// Elementwise / reduction kernels
// ---------------------------------------------------------------------------
__global__ __launch_bounds__(256) void k_add_pe(const float* __restrict__ src,
    float* __restrict__ xF, bf16* __restrict__ xB)
{
  int row = blockIdx.x, tid = threadIdx.x;
#pragma unroll
  for (int i = 0; i < 4; ++i) {
    int c  = tid + i * 256;
    int i2 = c & ~1;
    float freq = __expf(-logf(10000.f) * (float)i2 * (1.f / 1024.f));
    float ang  = (float)row * freq;
    float pe   = (c & 1) ? cosf(ang) : sinf(ang);
    float v = src[(size_t)row * 1024 + c] + pe;
    xF[(size_t)row * 1024 + c] = v;
    xB[(size_t)row * 1024 + c] = (bf16)v;
  }
}

__global__ __launch_bounds__(256) void k_add_ln(
    const float* __restrict__ x, const float* __restrict__ y,
    const float* __restrict__ g, const float* __restrict__ b,
    float* __restrict__ outF, bf16* __restrict__ outB)
{
  __shared__ float rs[256], rq[256];
  int row = blockIdx.x, tid = threadIdx.x;
  float v[4]; float s = 0.f, q = 0.f;
#pragma unroll
  for (int i = 0; i < 4; ++i) {
    int c = tid + i * 256;
    float t = x[(size_t)row*1024 + c] + y[(size_t)row*1024 + c];
    v[i] = t; s += t; q += t * t;
  }
  rs[tid] = s; rq[tid] = q; __syncthreads();
  for (int st = 128; st > 0; st >>= 1) {
    if (tid < st) { rs[tid] += rs[tid+st]; rq[tid] += rq[tid+st]; }
    __syncthreads();
  }
  float mean = rs[0] * (1.f / 1024.f);
  float var  = rq[0] * (1.f / 1024.f) - mean * mean;
  float rstd = rsqrtf(var + 1e-5f);
#pragma unroll
  for (int i = 0; i < 4; ++i) {
    int c = tid + i * 256;
    float t = (v[i] - mean) * rstd * g[c] + b[c];
    outF[(size_t)row*1024 + c] = t;
    outB[(size_t)row*1024 + c] = (bf16)t;
  }
}

__global__ __launch_bounds__(256) void k_tanh_dot(
    const float* __restrict__ t, const float* __restrict__ v,
    float* __restrict__ sOut)
{
  __shared__ float rs[256];
  int row = blockIdx.x, tid = threadIdx.x;
  float a = 0.f;
#pragma unroll
  for (int i = 0; i < 4; ++i) {
    int c = tid + i * 256;
    a += tanhf(t[(size_t)row*1024 + c]) * v[c];
  }
  rs[tid] = a; __syncthreads();
  for (int st = 128; st > 0; st >>= 1) {
    if (tid < st) rs[tid] += rs[tid+st];
    __syncthreads();
  }
  if (tid == 0) sOut[row] = rs[0];
}

__global__ __launch_bounds__(256) void k_fuse2(
    const float* __restrict__ s0, const float* __restrict__ s1,
    const float* __restrict__ h0, const float* __restrict__ h1,
    float* __restrict__ outF, bf16* __restrict__ outB)
{
  int row = blockIdx.x, tid = threadIdx.x;
  float a = s0[row], b = s1[row];
  float m  = fmaxf(a, b);
  float e0 = __expf(a - m), e1 = __expf(b - m);
  float inv = 1.f / (e0 + e1);
  float w0 = e0 * inv, w1 = e1 * inv;
#pragma unroll
  for (int i = 0; i < 4; ++i) {
    size_t idx = (size_t)row * 1024 + tid + i * 256;
    float t = w0 * h0[idx] + w1 * h1[idx];
    outF[idx] = t;
    if (outB) outB[idx] = (bf16)t;
  }
}

__global__ __launch_bounds__(256) void k_cls(
    const float* __restrict__ x, const float* __restrict__ Wc,
    const float* __restrict__ bc, float* __restrict__ out)
{
  __shared__ float red[256];
  __shared__ float lg[7];
  int row = blockIdx.x, tid = threadIdx.x;
  float acc[7] = {0,0,0,0,0,0,0};
#pragma unroll
  for (int i = 0; i < 4; ++i) {
    int d = tid + i * 256;
    float xv = x[(size_t)row*1024 + d];
#pragma unroll
    for (int c = 0; c < 7; ++c) acc[c] += xv * Wc[d*7 + c];
  }
  for (int c = 0; c < 7; ++c) {
    red[tid] = acc[c]; __syncthreads();
    for (int st = 128; st > 0; st >>= 1) {
      if (tid < st) red[tid] += red[tid+st];
      __syncthreads();
    }
    if (tid == 0) lg[c] = red[0] + bc[c];
    __syncthreads();
  }
  if (tid == 0) {
    float m = lg[0];
    for (int c = 1; c < 7; ++c) m = fmaxf(m, lg[c]);
    float s = 0.f;
    for (int c = 0; c < 7; ++c) s += __expf(lg[c] - m);
    float ls = logf(s);
    for (int c = 0; c < 7; ++c) out[row*7 + c] = lg[c] - m - ls;
  }
}

// ---------------------------------------------------------------------------
extern "C" void kernel_launch(void* const* d_in, const int* in_sizes, int n_in,
                              void* d_out, int out_size, void* d_ws, size_t ws_size,
                              hipStream_t stream)
{
  (void)in_sizes; (void)n_in; (void)out_size; (void)ws_size;
  const float* src  = (const float*)d_in[0];
  const int*   utt  = (const int*)d_in[1];
  const int*   spk  = (const int*)d_in[2];
  const int*   win  = (const int*)d_in[3];
  const float* Wqkv = (const float*)d_in[4];
  const float* bqkv = (const float*)d_in[5];
  const float* Wo   = (const float*)d_in[6];
  const float* bo   = (const float*)d_in[7];
  const float* ln1g = (const float*)d_in[8];
  const float* ln1b = (const float*)d_in[9];
  const float* Wf1  = (const float*)d_in[10];
  const float* bf1  = (const float*)d_in[11];
  const float* Wf2  = (const float*)d_in[12];
  const float* bf2  = (const float*)d_in[13];
  const float* ln2g = (const float*)d_in[14];
  const float* ln2b = (const float*)d_in[15];
  const float* f1W  = (const float*)d_in[16];
  const float* f1b  = (const float*)d_in[17];
  const float* f1v  = (const float*)d_in[18];
  const float* f2W  = (const float*)d_in[19];
  const float* f2b  = (const float*)d_in[20];
  const float* f2v  = (const float*)d_in[21];
  const float* clsW = (const float*)d_in[22];
  const float* clsb = (const float*)d_in[23];

  char* base = (char*)d_ws;
  size_t off = 0;
  auto alloc = [&](size_t bytes) -> char* {
    char* r = base + off;
    off += (bytes + 255) & ~(size_t)255;
    return r;
  };
  const size_t LD = (size_t)1024 * 1024;
  float* x0F = (float*)alloc(LD * 4);
  bf16*  x0B = (bf16*) alloc(LD * 2);
  float* xF  = (float*)alloc(LD * 4);
  bf16*  xB  = (bf16*) alloc(LD * 2);
  bf16*  qkvB = (bf16*)alloc((size_t)1024 * 3072 * 2);
  bf16*  oB   = (bf16*)alloc(LD * 2);
  float* yF   = (float*)alloc((size_t)1024 * 4096 * 4);
  bf16*  hB   = (bf16*) alloc((size_t)1024 * 4096 * 2);
  float* brF[3]; bf16* brB[3];
  for (int b = 0; b < 3; ++b) {
    brF[b] = (float*)alloc(LD * 4);
    brB[b] = (bf16*) alloc(LD * 2);
  }
  float* spF  = (float*)alloc(LD * 4);
  bf16*  spB  = (bf16*) alloc(LD * 2);
  float* fuF  = (float*)alloc(LD * 4);
  float* s0   = (float*)alloc(1024 * 4);
  float* s1   = (float*)alloc(1024 * 4);

  k_add_pe<<<1024, 256, 0, stream>>>(src, x0F, x0B);

  for (int b = 0; b < 3; ++b) {
    hipMemcpyAsync(xF, x0F, LD * 4, hipMemcpyDeviceToDevice, stream);
    hipMemcpyAsync(xB, x0B, LD * 2, hipMemcpyDeviceToDevice, stream);
    for (int l = 0; l < 3; ++l) {
      size_t wi = (size_t)b * 3 + l;
      // qkv = x @ Wqkv + bqkv  (bf16 out)
      k_gemm<false,false,true><<<dim3(24, 8), 256, 0, stream>>>(
          xB, Wqkv + wi * (size_t)1024 * 3072, bqkv + wi * 3072,
          nullptr, qkvB, 1024, 3072, 1024);
      // masked attention
      k_attn<<<dim3(16, 16), 128, 0, stream>>>(qkvB, oB, utt, spk, win, b);
      // y = o @ Wo + bo  (f32 out)
      k_gemm<false,true,false><<<dim3(8, 8), 256, 0, stream>>>(
          oB, Wo + wi * (size_t)1024 * 1024, bo + wi * 1024,
          yF, nullptr, 1024, 1024, 1024);
      k_add_ln<<<1024, 256, 0, stream>>>(xF, yF, ln1g + wi*1024, ln1b + wi*1024, xF, xB);
      // h = relu(x @ Wf1 + bf1)  (bf16 out)
      k_gemm<true,false,true><<<dim3(32, 8), 256, 0, stream>>>(
          xB, Wf1 + wi * (size_t)1024 * 4096, bf1 + wi * 4096,
          nullptr, hB, 1024, 4096, 1024);
      // f = h @ Wf2 + bf2  (f32 out)
      k_gemm<false,true,false><<<dim3(8, 8), 256, 0, stream>>>(
          hB, Wf2 + wi * (size_t)4096 * 1024, bf2 + wi * 1024,
          yF, nullptr, 1024, 1024, 4096);
      k_add_ln<<<1024, 256, 0, stream>>>(xF, yF, ln2g + wi*1024, ln2b + wi*1024, xF, xB);
    }
    hipMemcpyAsync(brF[b], xF, LD * 4, hipMemcpyDeviceToDevice, stream);
    hipMemcpyAsync(brB[b], xB, LD * 2, hipMemcpyDeviceToDevice, stream);
  }

  // fusion 1: sp = fuse(sm=br1, om=br2)
  k_gemm<false,true,false><<<dim3(8, 8), 256, 0, stream>>>(brB[1], f1W, f1b, yF, nullptr, 1024, 1024, 1024);
  k_tanh_dot<<<1024, 256, 0, stream>>>(yF, f1v, s0);
  k_gemm<false,true,false><<<dim3(8, 8), 256, 0, stream>>>(brB[2], f1W, f1b, yF, nullptr, 1024, 1024, 1024);
  k_tanh_dot<<<1024, 256, 0, stream>>>(yF, f1v, s1);
  k_fuse2<<<1024, 256, 0, stream>>>(s0, s1, brF[1], brF[2], spF, spB);

  // fusion 2: out = fuse(ct=br0, sp)
  k_gemm<false,true,false><<<dim3(8, 8), 256, 0, stream>>>(brB[0], f2W, f2b, yF, nullptr, 1024, 1024, 1024);
  k_tanh_dot<<<1024, 256, 0, stream>>>(yF, f2v, s0);
  k_gemm<false,true,false><<<dim3(8, 8), 256, 0, stream>>>(spB, f2W, f2b, yF, nullptr, 1024, 1024, 1024);
  k_tanh_dot<<<1024, 256, 0, stream>>>(yF, f2v, s1);
  k_fuse2<<<1024, 256, 0, stream>>>(s0, s1, brF[0], spF, fuF, nullptr);

  // classifier + log_softmax
  k_cls<<<1024, 256, 0, stream>>>(fuF, clsW, clsb, (float*)d_out);
}
